// L2_85023172591652
// MI455X (gfx1250) — compile-verified, matrix operand order checked by reading, not run on hardware
//
#include <hip/hip_runtime.h>

// ---------------------------------------------------------------------------
// Problem constants (match reference)
// ---------------------------------------------------------------------------
constexpr int B_ROWS = 2048;   // batch
constexpr int D_DIM  = 1024;   // feature dim (K)
constexpr int C_CENT = 8192;   // centroids (N)

constexpr int BM      = 32;    // rows per block (2 M-tiles)
constexpr int CHUNK_N = 256;   // logits chunk width per pass (16 N-tiles)
constexpr int CBLK    = 1024;  // centroid columns per block
constexpr int NCHUNKS = C_CENT / CBLK;   // 8 column-blocks -> 8 partials/row

// ---------------------------------------------------------------------------
// Types for WMMA (gfx1250, wave32)
// ---------------------------------------------------------------------------
typedef __attribute__((ext_vector_type(8)))  __bf16 v8bf;
typedef __attribute__((ext_vector_type(16))) __bf16 v16bf;
typedef __attribute__((ext_vector_type(8)))  float  v8f;

// Load a 16-element bf16 fragment for one lane:
//   elements 0..7  = K offsets [0..7]   at p
//   elements 8..15 = K offsets [16..23] at p+16
// (per CDNA5 16-bit A/B VGPR layout; caller applies the lane>=16 "+8" K shift)
__device__ __forceinline__ v16bf ld_frag(const __bf16* p) {
    v8bf a = *(const v8bf*)p;
    v8bf b = *(const v8bf*)(p + 16);
    v16bf r;
#pragma unroll
    for (int i = 0; i < 8; ++i) { r[i] = a[i]; r[i + 8] = b[i]; }
    return r;
}

__device__ __forceinline__ v8f wmma_bf16(v16bf a, v16bf b, v8f c) {
    return __builtin_amdgcn_wmma_f32_16x16x32_bf16(
        /*neg_a=*/false, a, /*neg_b=*/false, b,
        /*c_mod=*/(short)0, c, /*reuse_a=*/false, /*reuse_b=*/false);
}

// ---------------------------------------------------------------------------
// Kernel 0: zero the 2-float output (loss, score)
// ---------------------------------------------------------------------------
__global__ void zero_out_kernel(float* out) {
    if (threadIdx.x < 2) out[threadIdx.x] = 0.0f;
}

// ---------------------------------------------------------------------------
// Kernel 1: split x (f32) into bf16 hi/lo
// ---------------------------------------------------------------------------
__global__ __launch_bounds__(256) void prep_x_kernel(
    const float* __restrict__ x, __bf16* __restrict__ xh, __bf16* __restrict__ xl) {
    int i = blockIdx.x * 256 + threadIdx.x;
    float v  = x[i];
    __bf16 h = (__bf16)v;
    xh[i] = h;
    xl[i] = (__bf16)(v - (float)h);
}

// ---------------------------------------------------------------------------
// Kernel 2: split centroids into bf16 hi/lo + row norms c2
// ---------------------------------------------------------------------------
__global__ __launch_bounds__(256) void prep_c_kernel(
    const float* __restrict__ c, __bf16* __restrict__ ch, __bf16* __restrict__ cl,
    float* __restrict__ c2) {
    __shared__ float red[256];
    int row = blockIdx.x, tid = threadIdx.x;
    const float* cr = c + (size_t)row * D_DIM;
    float acc = 0.0f;
    for (int j = tid; j < D_DIM; j += 256) {
        float v  = cr[j];
        __bf16 h = (__bf16)v;
        ch[(size_t)row * D_DIM + j] = h;
        cl[(size_t)row * D_DIM + j] = (__bf16)(v - (float)h);
        acc = fmaf(v, v, acc);
    }
    red[tid] = acc;
    __syncthreads();
    for (int s = 128; s > 0; s >>= 1) {
        if (tid < s) red[tid] += red[tid + s];
        __syncthreads();
    }
    if (tid == 0) c2[row] = red[0];
}

// ---------------------------------------------------------------------------
// Kernel 3: fused GEMM (bf16x3 split, WMMA) + online softmax/argmax partials.
// Block = 256 threads (8 waves). Block tile: 32 rows x 1024 centroids,
// processed as 4 passes of 32x256 logits staged in LDS.
// Each wave owns a 2(M) x 2(N) register tile of 16x16 accumulators:
//   per k-step: 8 A-frag + 8 B-frag b128 loads feed 12 WMMAs (ratio 1.33).
// ---------------------------------------------------------------------------
__global__ __launch_bounds__(256) void gemm_softmax_kernel(
    const __bf16* __restrict__ xh, const __bf16* __restrict__ xl,
    const __bf16* __restrict__ ch, const __bf16* __restrict__ cl,
    const float*  __restrict__ c2, float4* __restrict__ partials) {

    __shared__ float lg[BM][CHUNK_N + 4];     // logits chunk (33.3 KB)
    __shared__ float red_m[BM][8];
    __shared__ float red_s[BM][8];
    __shared__ int   red_i[BM][8];
    __shared__ float rowM[BM], rowS[BM];
    __shared__ int   rowI[BM];
    __shared__ float chM[BM];
    __shared__ int   chI[BM];

    const int tid  = threadIdx.x;
    const int wave = tid >> 5;
    const int lane = tid & 31;
    const int cblock = blockIdx.x;            // 0..7
    const int mblock = blockIdx.y;            // 0..63
    const int mbase  = mblock * BM;
    const int cbase  = cblock * CBLK;

    if (tid < BM) { rowM[tid] = -3.4e38f; rowS[tid] = 0.0f; rowI[tid] = 0; }
    __syncthreads();

    const int ntp   = wave * 2;               // first of 2 owned 16-col tiles (0..15)
    const int lrow  = lane & 15;
    const int khalf = (lane >> 4) ? 8 : 0;    // lane-half K shift

    // A pointers for both 16-row tiles (all waves share the same 32 rows -> WGP$)
    const __bf16* a0h_p = xh + (size_t)(mbase + lrow) * D_DIM;
    const __bf16* a0l_p = xl + (size_t)(mbase + lrow) * D_DIM;
    const __bf16* a1h_p = a0h_p + (size_t)16 * D_DIM;
    const __bf16* a1l_p = a0l_p + (size_t)16 * D_DIM;

    for (int chunk = 0; chunk < CBLK / CHUNK_N; ++chunk) {
        const int nchunkbase = cbase + chunk * CHUNK_N;
        v8f acc00 = {}; v8f acc01 = {};       // mt0 x {nt0, nt1}
        v8f acc10 = {}; v8f acc11 = {};       // mt1 x {nt0, nt1}

        const __bf16* b0h = ch + (size_t)(nchunkbase + ntp * 16 + lrow) * D_DIM;
        const __bf16* b0l = cl + (size_t)(nchunkbase + ntp * 16 + lrow) * D_DIM;
        const __bf16* b1h = b0h + (size_t)16 * D_DIM;
        const __bf16* b1l = b0l + (size_t)16 * D_DIM;

#pragma unroll 4
        for (int k0 = 0; k0 < D_DIM; k0 += 32) {
            const int ka = k0 + khalf;
            v16bf a0h = ld_frag(a0h_p + ka);
            v16bf a0l = ld_frag(a0l_p + ka);
            v16bf a1h = ld_frag(a1h_p + ka);
            v16bf a1l = ld_frag(a1l_p + ka);

            v16bf bh = ld_frag(b0h + ka);
            v16bf bl = ld_frag(b0l + ka);
            acc00 = wmma_bf16(a0h, bh, acc00);
            acc00 = wmma_bf16(a0h, bl, acc00);
            acc00 = wmma_bf16(a0l, bh, acc00);
            acc10 = wmma_bf16(a1h, bh, acc10);
            acc10 = wmma_bf16(a1h, bl, acc10);
            acc10 = wmma_bf16(a1l, bh, acc10);

            bh = ld_frag(b1h + ka);
            bl = ld_frag(b1l + ka);
            acc01 = wmma_bf16(a0h, bh, acc01);
            acc01 = wmma_bf16(a0h, bl, acc01);
            acc01 = wmma_bf16(a0l, bh, acc01);
            acc11 = wmma_bf16(a1h, bh, acc11);
            acc11 = wmma_bf16(a1h, bl, acc11);
            acc11 = wmma_bf16(a1l, bh, acc11);
        }

        // Epilogue: logits = 2*dot - c2  (||x||^2 cancels in softmax/argmax)
        const int nc0  = ntp * 16 + lrow;
        const int nc1  = nc0 + 16;
        const float c2v0 = c2[nchunkbase + nc0];
        const float c2v1 = c2[nchunkbase + nc1];
        const int mrow = (lane >> 4) << 3;
#pragma unroll
        for (int r = 0; r < 8; ++r) {
            lg[mrow + r][nc0]      = 2.0f * acc00[r] - c2v0;
            lg[mrow + r][nc1]      = 2.0f * acc01[r] - c2v1;
            lg[16 + mrow + r][nc0] = 2.0f * acc10[r] - c2v0;
            lg[16 + mrow + r][nc1] = 2.0f * acc11[r] - c2v1;
        }
        __syncthreads();

        // Per-row chunk reduction: 8 threads per row, 32 cols each.
        const int rr = tid >> 3, rs = tid & 7;
        float lm = -3.4e38f; int li = 0;
#pragma unroll
        for (int j = 0; j < 32; ++j) {
            int   col = rs * 32 + j;
            float v   = lg[rr][col];
            if (v > lm) { lm = v; li = col; }
        }
        red_m[rr][rs] = lm; red_i[rr][rs] = li;
        __syncthreads();

        if (rs == 0) {
            float cm = red_m[rr][0]; int ci = red_i[rr][0];
#pragma unroll
            for (int j = 1; j < 8; ++j)
                if (red_m[rr][j] > cm) { cm = red_m[rr][j]; ci = red_i[rr][j]; }
            chM[rr] = cm; chI[rr] = ci;
        }
        __syncthreads();

        const float cm = chM[rr];
        float ls = 0.0f;
#pragma unroll
        for (int j = 0; j < 32; ++j) ls += __expf(lg[rr][rs * 32 + j] - cm);
        red_s[rr][rs] = ls;
        __syncthreads();

        if (rs == 0) {
            float cs = 0.0f;
#pragma unroll
            for (int j = 0; j < 8; ++j) cs += red_s[rr][j];
            float M0 = rowM[rr], S0 = rowS[rr];
            float Mn = fmaxf(M0, cm);
            rowS[rr] = S0 * __expf(M0 - Mn) + cs * __expf(cm - Mn);
            if (cm > M0) rowI[rr] = nchunkbase + chI[rr];
            rowM[rr] = Mn;
        }
        __syncthreads();
    }

    if (tid < BM) {
        int mg = mbase + tid;
        partials[(size_t)mg * NCHUNKS + cblock] =
            make_float4(rowM[tid], rowS[tid], __int_as_float(rowI[tid]), 0.0f);
    }
}

// ---------------------------------------------------------------------------
// Kernel 4: combine partials per row (1 wave / row), exact f32 target logit,
// atomic accumulation of loss and accuracy.
// ---------------------------------------------------------------------------
__global__ __launch_bounds__(256) void combine_kernel(
    const float* __restrict__ x, const int* __restrict__ y,
    const float* __restrict__ cent, const float* __restrict__ c2,
    const float4* __restrict__ partials, float* __restrict__ out) {

    const int wave = threadIdx.x >> 5;
    const int lane = threadIdx.x & 31;
    const int b    = blockIdx.x * 8 + wave;

    float ml = -3.4e38f, sl = 0.0f; int il = 0;
    if (lane < NCHUNKS) {
        float4 p = partials[(size_t)b * NCHUNKS + lane];
        ml = p.x; sl = p.y; il = __float_as_int(p.z);
    }

    // all-reduce max + argmax across the wave
    float M = ml; int I = il;
#pragma unroll
    for (int off = 16; off > 0; off >>= 1) {
        float om = __shfl_xor(M, off, 32);
        int   oi = __shfl_xor(I, off, 32);
        if (om > M) { M = om; I = oi; }
    }
    // rescaled sum-exp
    float seff = sl * __expf(ml - M);
#pragma unroll
    for (int off = 16; off > 0; off >>= 1) seff += __shfl_xor(seff, off, 32);

    // exact f32 target logit: 2*x_b . c_y - ||c_y||^2
    const int yv = y[b];
    const float* xr = x    + (size_t)b  * D_DIM;
    const float* cr = cent + (size_t)yv * D_DIM;
    float d = 0.0f;
    for (int k = lane; k < D_DIM; k += 32) d = fmaf(xr[k], cr[k], d);
#pragma unroll
    for (int off = 16; off > 0; off >>= 1) d += __shfl_xor(d, off, 32);

    if (lane == 0) {
        float logit_y = 2.0f * d - c2[yv];
        float loss    = (M + __logf(seff)) - logit_y;
        atomicAdd(out + 0, loss * (1.0f / (float)B_ROWS));
        atomicAdd(out + 1, (I == yv) ? (1.0f / (float)B_ROWS) : 0.0f);
    }
}

// ---------------------------------------------------------------------------
// Launcher
// ---------------------------------------------------------------------------
extern "C" void kernel_launch(void* const* d_in, const int* in_sizes, int n_in,
                              void* d_out, int out_size, void* d_ws, size_t ws_size,
                              hipStream_t stream) {
    const float* x    = (const float*)d_in[0];
    const int*   y    = (const int*)  d_in[1];
    const float* cent = (const float*)d_in[2];
    float*       out  = (float*)d_out;

    char* ws = (char*)d_ws;
    size_t off = 0;
    __bf16* xh = (__bf16*)(ws + off); off += (size_t)B_ROWS * D_DIM * 2;
    __bf16* xl = (__bf16*)(ws + off); off += (size_t)B_ROWS * D_DIM * 2;
    __bf16* ch = (__bf16*)(ws + off); off += (size_t)C_CENT * D_DIM * 2;
    __bf16* cl = (__bf16*)(ws + off); off += (size_t)C_CENT * D_DIM * 2;
    float*  c2 = (float*) (ws + off); off += (size_t)C_CENT * 4;
    float4* pp = (float4*)(ws + off); off += (size_t)B_ROWS * NCHUNKS * 16;

    zero_out_kernel<<<1, 32, 0, stream>>>(out);
    prep_x_kernel<<<(B_ROWS * D_DIM) / 256, 256, 0, stream>>>(x, xh, xl);
    prep_c_kernel<<<C_CENT, 256, 0, stream>>>(cent, ch, cl, c2);

    dim3 grid(NCHUNKS, B_ROWS / BM);   // 8 x 64 = 512 blocks
    gemm_softmax_kernel<<<grid, 256, 0, stream>>>(xh, xl, ch, cl, c2, pp);

    combine_kernel<<<B_ROWS / 8, 256, 0, stream>>>(x, y, cent, c2, pp, out);
}
